// MultiTaskFEGIN_15779709845720
// MI455X (gfx1250) — matrix operand order repaired and checked by reading, not compiled, not run on hardware
//
#include <hip/hip_runtime.h>

// ---------------- problem constants ----------------
#define NN   100000   // nodes
#define EE   1200000  // edges
#define FF   32       // input features
#define HH   64       // hidden
#define GG   256      // graphs
#define CC   200000   // candidates
#define NCLS 10       // classes
#define DD   192      // L*H concat dim

typedef __attribute__((ext_vector_type(16))) __bf16 v16bf;
typedef __attribute__((ext_vector_type(8)))  float  v8f;

union FragBF  { v16bf v; unsigned short u[16]; uint4 q[2]; };
union FragAcc { v8f   v; float          f[8];  };

__device__ __forceinline__ unsigned short f2bf(float f) {
  unsigned int u = __float_as_uint(f);
  u += 0x7fffu + ((u >> 16) & 1u);      // round-to-nearest-even
  return (unsigned short)(u >> 16);
}

// A fragment: 16 rows x 32 K, bf16, row-major LDS tile.
// Per ISA: lane<16 -> row=lane, K in {8h..8h+7, 16+8h..23+8h}; two 16B groups.
// Requires: base 16B-aligned, stride multiple of 8 ushorts.
__device__ __forceinline__ v16bf load_frag_a(const unsigned short* base, int stride, int lane) {
  FragBF f;
  const int half = lane >> 4, r = lane & 15;
  const unsigned short* p = base + r * stride + 8 * half;
  f.q[0] = *(const uint4*)(p);        // K = 8*half .. 8*half+7
  f.q[1] = *(const uint4*)(p + 16);   // K = 16+8*half .. 23+8*half
  return f.v;
}

// B fragment from pre-swizzled 512-ushort tile: lane's 16 values contiguous.
__device__ __forceinline__ v16bf load_frag_b_swz(const unsigned short* tile, int lane) {
  FragBF f;
  const unsigned short* p = tile + lane * 16;
  f.q[0] = *(const uint4*)(p);
  f.q[1] = *(const uint4*)(p + 8);
  return f.v;
}

__device__ __forceinline__ v8f wmma_bf16(v16bf a, v16bf b, v8f c) {
  return __builtin_amdgcn_wmma_f32_16x16x32_bf16(false, a, false, b, (short)0, c, false, false);
}

// ---------------- utility kernels ----------------
__global__ void k_zero(float* p, int n) {
  int i = blockIdx.x * blockDim.x + threadIdx.x;
  if (i < n) p[i] = 0.f;
}

// convert W[K][Nn] f32 -> bf16 in WMMA-B-fragment-swizzled layout:
// dst[((ks*(Nn/16)+nt)*512) + lane*16 + j] = bf16(W[(ks*32 + (j>>3)*16 + (lane>>4)*8 + (j&7))*Nn + nt*16 + (lane&15)])
__global__ void k_f2bf_swz(const float* __restrict__ W, unsigned short* __restrict__ dst,
                           int K, int Nn) {
  int d = blockIdx.x * blockDim.x + threadIdx.x;
  if (d >= K * Nn) return;
  int tile = d >> 9, within = d & 511;
  int lane = within >> 4, j = within & 15;
  int ntCnt = Nn >> 4;
  int nt = tile % ntCnt, ks = tile / ntCnt;
  int k = ks * 32 + ((j >> 3) << 4) + ((lane >> 4) << 3) + (j & 7);
  int n = (nt << 4) + (lane & 15);
  dst[d] = f2bf(W[k * Nn + n]);
}

// agg[dst] += h[src] ; F-wide rows, F = 1<<fshift
__global__ void k_edge_scatter(const int* __restrict__ src, const int* __restrict__ dst,
                               const float* __restrict__ h, int ld,
                               float* __restrict__ agg, int fshift, int total) {
  int i = blockIdx.x * blockDim.x + threadIdx.x;
  if (i >= total) return;
  int F = 1 << fshift;
  int e = i >> fshift, f = i & (F - 1);
  atomicAdd(&agg[dst[e] * F + f], h[src[e] * ld + f]);
}

// ---------------- fused GIN MLP: relu(relu((1+eps)x+agg @W1+b1)@W2+b2) -> BN ----------------
#define SA_LD 80   // ushorts; 160B row -> 16B aligned groups

template <int F_IN, int FSHIFT>
__global__ void __launch_bounds__(256) k_gin_mlp(
    const float* __restrict__ in, int ld_in,
    const float* __restrict__ agg, const float* __restrict__ epsp,
    const unsigned short* __restrict__ W1s, const float* __restrict__ b1,
    const unsigned short* __restrict__ W2s, const float* __restrict__ b2,
    const float* __restrict__ bn_g, const float* __restrict__ bn_b,
    const float* __restrict__ bn_m, const float* __restrict__ bn_v,
    float* __restrict__ out, int ld_out) {
  __shared__ __align__(16) unsigned short sA[64 * SA_LD];
  __shared__ __align__(16) unsigned short sW1[F_IN * 64];   // swizzled tiles
  __shared__ __align__(16) unsigned short sH[64 * SA_LD];
  __shared__ __align__(16) unsigned short sW2[64 * 64];     // swizzled tiles

  const int tid = threadIdx.x;
  const int n0  = blockIdx.x * 64;
  const float eps1 = 1.0f + epsp[0];

  for (int i = tid; i < 64 * F_IN; i += 256) {
    int r = i >> FSHIFT, f = i & (F_IN - 1);
    int node = n0 + r;
    float v = 0.f;
    if (node < NN) v = eps1 * in[node * ld_in + f] + agg[node * F_IN + f];
    sA[r * SA_LD + f] = f2bf(v);
  }
  for (int i = tid; i < F_IN * 64; i += 256) sW1[i] = W1s[i];
  for (int i = tid; i < 64 * 64;  i += 256) sW2[i] = W2s[i];
  __syncthreads();

  const int wave = tid >> 5, lane = tid & 31;
  const int half = lane >> 4, nl = lane & 15;

  // GEMM1: [64,F_IN] x [F_IN,64] -> relu -> sH (bf16)
  for (int t = wave; t < 16; t += 8) {
    int mt = t >> 2, nt = t & 3;
    FragAcc acc;
#pragma unroll
    for (int i = 0; i < 8; ++i) acc.f[i] = 0.f;
#pragma unroll
    for (int kk = 0; kk < F_IN; kk += 32) {
      v16bf a = load_frag_a(&sA[mt * 16 * SA_LD + kk], SA_LD, lane);
      v16bf b = load_frag_b_swz(&sW1[((kk >> 5) * 4 + nt) * 512], lane);
      acc.v = wmma_bf16(a, b, acc.v);
    }
    int j = nt * 16 + nl;
    float bj = b1[j];
#pragma unroll
    for (int i = 0; i < 8; ++i) {
      float v = acc.f[i] + bj;
      sH[(mt * 16 + i + 8 * half) * SA_LD + j] = f2bf(v > 0.f ? v : 0.f);
    }
  }
  __syncthreads();

  // GEMM2: [64,64] x [64,64] -> relu -> BN -> out (f32)
  for (int t = wave; t < 16; t += 8) {
    int mt = t >> 2, nt = t & 3;
    FragAcc acc;
#pragma unroll
    for (int i = 0; i < 8; ++i) acc.f[i] = 0.f;
#pragma unroll
    for (int kk = 0; kk < 64; kk += 32) {
      v16bf a = load_frag_a(&sH[mt * 16 * SA_LD + kk], SA_LD, lane);
      v16bf b = load_frag_b_swz(&sW2[((kk >> 5) * 4 + nt) * 512], lane);
      acc.v = wmma_bf16(a, b, acc.v);
    }
    int j = nt * 16 + nl;
    float bj = b2[j], g = bn_g[j], be = bn_b[j], mu = bn_m[j];
    float iv = rsqrtf(bn_v[j] + 1e-5f);
#pragma unroll
    for (int i = 0; i < 8; ++i) {
      int node = n0 + mt * 16 + i + 8 * half;
      if (node < NN) {
        float v = acc.f[i] + bj;
        v = v > 0.f ? v : 0.f;
        out[node * ld_out + j] = g * (v - mu) * iv + be;
      }
    }
  }
}

// ---------------- pooling ----------------
__global__ void k_pool_scatter(const float* __restrict__ ne, const int* __restrict__ batch,
                               float* __restrict__ gsum, float* __restrict__ gcnt, int total) {
  int i = blockIdx.x * blockDim.x + threadIdx.x;
  if (i >= total) return;
  int n = i / DD, d = i - n * DD;
  int g = batch[n];
  atomicAdd(&gsum[g * DD + d], ne[i]);
  if (d == 0) atomicAdd(&gcnt[g], 1.0f);
}

__global__ void k_pool_final(const float* __restrict__ gsum, const float* __restrict__ gcnt,
                             float* __restrict__ ge) {
  int i = blockIdx.x * blockDim.x + threadIdx.x;
  if (i >= GG * DD) return;
  int g = i / DD;
  float c = gcnt[g];
  ge[i] = gsum[i] / (c < 1.f ? 1.f : c);
}

// ---------------- classification head (tiny, f32) ----------------
__global__ void __launch_bounds__(128) k_class_head(
    const float* __restrict__ ge,
    const float* __restrict__ W1, const float* __restrict__ b1,
    const float* __restrict__ W2, const float* __restrict__ b2,
    const float* __restrict__ W3, const float* __restrict__ b3,
    const float* __restrict__ W4, const float* __restrict__ b4,
    float* __restrict__ out_logsm, int* __restrict__ comp_idx) {
  __shared__ float s_in[DD], s1[128], s2[64], s3[64], sl[NCLS];
  const int g = blockIdx.x, tid = threadIdx.x;
  for (int i = tid; i < DD; i += 128) s_in[i] = ge[g * DD + i];
  __syncthreads();
  { float a = b1[tid]; for (int k = 0; k < DD; ++k) a += s_in[k] * W1[k * 128 + tid];
    s1[tid] = a > 0.f ? a : 0.f; }
  __syncthreads();
  if (tid < 64) { float a = b2[tid]; for (int k = 0; k < 128; ++k) a += s1[k] * W2[k * 64 + tid];
    s2[tid] = a > 0.f ? a : 0.f; }
  __syncthreads();
  if (tid < 64) { float a = b3[tid]; for (int k = 0; k < 64; ++k) a += s2[k] * W3[k * 64 + tid];
    s3[tid] = a > 0.f ? a : 0.f; }
  __syncthreads();
  if (tid < NCLS) { float a = b4[tid]; for (int k = 0; k < 64; ++k) a += s3[k] * W4[k * NCLS + tid];
    sl[tid] = a; }
  __syncthreads();
  if (tid == 0) {
    float mx = sl[0]; int am = 0;
    for (int c = 1; c < NCLS; ++c) if (sl[c] > mx) { mx = sl[c]; am = c; }
    float se = 0.f;
    for (int c = 0; c < NCLS; ++c) se += __expf(sl[c] - mx);
    float ls = __logf(se);
    for (int c = 0; c < NCLS; ++c) out_logsm[g * NCLS + c] = sl[c] - mx - ls;
    comp_idx[g] = am;
  }
}

// ---------------- comp_emb @ ep_W1[:D] + b1 -> [10,128] ----------------
__global__ void k_comp_contrib(const float* __restrict__ comp_emb, const float* __restrict__ epW1,
                               const float* __restrict__ epb1, float* __restrict__ cc) {
  int nc = blockIdx.x, j = threadIdx.x;   // 10 blocks x 128
  float a = epb1[j];
  for (int k = 0; k < DD; ++k) a += comp_emb[nc * DD + k] * epW1[k * 128 + j];
  cc[nc * 128 + j] = a;
}

// ---------------- link prediction: 128 candidates per block ----------------
#define DE_LD 208   // ushorts; 416B row -> 16B aligned groups
#define H1_LD 136   // ushorts; 272B row -> 16B aligned groups

__global__ void __launch_bounds__(256, 1) k_link(
    const int* __restrict__ cand, const int* __restrict__ batch,
    const int* __restrict__ comp_idx, const float* __restrict__ ne,
    const float* __restrict__ cc,
    const unsigned short* __restrict__ W1s,   // swizzled bf16 ep_W1[192:384,:]
    const unsigned short* __restrict__ W2s,   // swizzled bf16 ep_W2
    const float* __restrict__ b2, const float* __restrict__ W3, const float* __restrict__ b3,
    float* __restrict__ out) {
  __shared__ __align__(16) unsigned short sDE[128 * DE_LD];  // 53248 B (recycled as f32 h2)
  __shared__ __align__(16) unsigned short sW1[DD * 128];     // 49152 B swizzled
  __shared__ __align__(16) unsigned short sH1[128 * H1_LD];  // 34816 B
  __shared__ __align__(16) unsigned short sW2[128 * 64];     // 16384 B swizzled
  __shared__ int sCls[128];
  float* sH2 = (float*)&sDE[0];               // stride 66 floats (33792 B <= 53248 B)

  const int tid = threadIdx.x;
  const int c0  = blockIdx.x * 128;

  // stage candidate node embeddings (2 threads per row)
  {
    int r = tid >> 1, part = tid & 1;
    int c = c0 + r;
    bool valid = (c < CC);
    int node = valid ? cand[c] : 0;
    const float* srcp = ne + node * DD;
    for (int f = part * 96; f < part * 96 + 96; ++f)
      sDE[r * DE_LD + f] = valid ? f2bf(srcp[f]) : (unsigned short)0;
    if (part == 0) sCls[r] = valid ? comp_idx[batch[node]] : 0;
  }
  for (int i = tid; i < DD * 128; i += 256) sW1[i] = W1s[i];
  for (int i = tid; i < 128 * 64; i += 256) sW2[i] = W2s[i];
  __syncthreads();

  const int wave = tid >> 5, lane = tid & 31;
  const int half = lane >> 4, nl = lane & 15;

  // GEMM1: [128,192] x [192,128], + comp_contrib[class], relu -> sH1 bf16
  for (int t = wave * 8; t < wave * 8 + 8; ++t) {
    int mt = t >> 3, nt = t & 7;
    FragAcc acc;
#pragma unroll
    for (int i = 0; i < 8; ++i) acc.f[i] = 0.f;
#pragma unroll
    for (int kk = 0; kk < DD; kk += 32) {
      v16bf a = load_frag_a(&sDE[mt * 16 * DE_LD + kk], DE_LD, lane);
      v16bf b = load_frag_b_swz(&sW1[((kk >> 5) * 8 + nt) * 512], lane);
      acc.v = wmma_bf16(a, b, acc.v);
    }
    int j = nt * 16 + nl;
#pragma unroll
    for (int i = 0; i < 8; ++i) {
      int r = mt * 16 + i + 8 * half;
      float v = acc.f[i] + cc[sCls[r] * 128 + j];
      sH1[r * H1_LD + j] = f2bf(v > 0.f ? v : 0.f);
    }
  }
  __syncthreads();

  // GEMM2: [128,128] x [128,64], +b2, relu -> sH2 f32 (recycled sDE)
  for (int t = wave * 4; t < wave * 4 + 4; ++t) {
    int mt = t >> 2, nt = t & 3;
    FragAcc acc;
#pragma unroll
    for (int i = 0; i < 8; ++i) acc.f[i] = 0.f;
#pragma unroll
    for (int kk = 0; kk < 128; kk += 32) {
      v16bf a = load_frag_a(&sH1[mt * 16 * H1_LD + kk], H1_LD, lane);
      v16bf b = load_frag_b_swz(&sW2[((kk >> 5) * 4 + nt) * 512], lane);
      acc.v = wmma_bf16(a, b, acc.v);
    }
    int j = nt * 16 + nl;
    float bj = b2[j];
#pragma unroll
    for (int i = 0; i < 8; ++i) {
      int r = mt * 16 + i + 8 * half;
      float v = acc.f[i] + bj;
      sH2[r * 66 + j] = v > 0.f ? v : 0.f;
    }
  }
  __syncthreads();

  // final 64-dot + sigmoid
  if (tid < 128) {
    int c = c0 + tid;
    if (c < CC) {
      float s = b3[0];
#pragma unroll 8
      for (int j = 0; j < 64; ++j) s += sH2[tid * 66 + j] * W3[j];
      out[GG * NCLS + c] = 1.f / (1.f + __expf(-s));
    }
  }
}

// ---------------- host side ----------------
static inline int cdiv(int a, int b) { return (a + b - 1) / b; }

extern "C" void kernel_launch(void* const* d_in, const int* in_sizes, int n_in,
                              void* d_out, int out_size, void* d_ws, size_t ws_size,
                              hipStream_t stream) {
  // inputs (setup_inputs order)
  const float* x        = (const float*)d_in[0];
  const int*   eidx     = (const int*)d_in[1];
  const int*   batch    = (const int*)d_in[2];
  const int*   cand     = (const int*)d_in[3];
  const float* c1_W1    = (const float*)d_in[4];
  const float* c1_b1    = (const float*)d_in[5];
  const float* c1_W2    = (const float*)d_in[6];
  const float* c1_b2    = (const float*)d_in[7];
  const float* c1_g     = (const float*)d_in[8];
  const float* c1_be    = (const float*)d_in[9];
  const float* c1_m     = (const float*)d_in[10];
  const float* c1_v     = (const float*)d_in[11];
  const float* c1_eps   = (const float*)d_in[12];
  const float* cs_W1    = (const float*)d_in[13];
  const float* cs_b1    = (const float*)d_in[14];
  const float* cs_W2    = (const float*)d_in[15];
  const float* cs_b2    = (const float*)d_in[16];
  const float* cs_g     = (const float*)d_in[17];
  const float* cs_be    = (const float*)d_in[18];
  const float* cs_m     = (const float*)d_in[19];
  const float* cs_v     = (const float*)d_in[20];
  const float* cs_eps   = (const float*)d_in[21];
  const float* nc_W1    = (const float*)d_in[22];
  const float* nc_b1    = (const float*)d_in[23];
  const float* nc_W2    = (const float*)d_in[24];
  const float* nc_b2    = (const float*)d_in[25];
  const float* nc_W3    = (const float*)d_in[26];
  const float* nc_b3    = (const float*)d_in[27];
  const float* nc_W4    = (const float*)d_in[28];
  const float* nc_b4    = (const float*)d_in[29];
  const float* comp_emb = (const float*)d_in[30];
  const float* ep_W1    = (const float*)d_in[31];
  const float* ep_b1    = (const float*)d_in[32];
  const float* ep_W2    = (const float*)d_in[33];
  const float* ep_b2    = (const float*)d_in[34];
  const float* ep_W3    = (const float*)d_in[35];
  const float* ep_b3    = (const float*)d_in[36];

  const int* e_src = eidx;
  const int* e_dst = eidx + EE;
  float* out = (float*)d_out;

  // workspace layout
  char* ws = (char*)d_ws;
  size_t off = 0;
  auto alloc = [&](size_t bytes) -> char* {
    char* p = ws + off;
    off = (off + bytes + 255) & ~(size_t)255;
    return p;
  };
  float* node_emb = (float*)alloc((size_t)NN * DD * 4);   // [N,192]
  float* agg      = (float*)alloc((size_t)NN * HH * 4);   // [N,<=64]
  float* gsum     = (float*)alloc((size_t)GG * DD * 4);
  float* gcnt     = (float*)alloc((size_t)GG * 4);
  float* gemb     = (float*)alloc((size_t)GG * DD * 4);
  int*   cidx     = (int*)  alloc((size_t)GG * 4);
  float* ccontrib = (float*)alloc((size_t)NCLS * 128 * 4);
  unsigned short* c1W1s = (unsigned short*)alloc((size_t)FF * HH * 2);
  unsigned short* c1W2s = (unsigned short*)alloc((size_t)HH * HH * 2);
  unsigned short* csW1s = (unsigned short*)alloc((size_t)2 * HH * HH * 2);
  unsigned short* csW2s = (unsigned short*)alloc((size_t)2 * HH * HH * 2);
  unsigned short* epW1s = (unsigned short*)alloc((size_t)DD * 128 * 2);
  unsigned short* epW2s = (unsigned short*)alloc((size_t)128 * 64 * 2);

  // 1) weight conversions: f32 -> bf16, pre-swizzled into WMMA-B fragment order
  k_f2bf_swz<<<cdiv(FF * HH, 256), 256, 0, stream>>>(c1_W1, c1W1s, FF, HH);
  k_f2bf_swz<<<cdiv(HH * HH, 256), 256, 0, stream>>>(c1_W2, c1W2s, HH, HH);
  for (int l = 0; l < 2; ++l) {
    k_f2bf_swz<<<cdiv(HH * HH, 256), 256, 0, stream>>>(cs_W1 + l * HH * HH, csW1s + l * HH * HH, HH, HH);
    k_f2bf_swz<<<cdiv(HH * HH, 256), 256, 0, stream>>>(cs_W2 + l * HH * HH, csW2s + l * HH * HH, HH, HH);
  }
  k_f2bf_swz<<<cdiv(DD * 128, 256), 256, 0, stream>>>(ep_W1 + DD * 128, epW1s, DD, 128);
  k_f2bf_swz<<<cdiv(128 * 64, 256), 256, 0, stream>>>(ep_W2, epW2s, 128, 64);

  // 2) GIN layer 1 (F_in=32, input = x)
  k_zero<<<cdiv(NN * FF, 256), 256, 0, stream>>>(agg, NN * FF);
  k_edge_scatter<<<cdiv(EE * FF, 256), 256, 0, stream>>>(e_src, e_dst, x, FF, agg, 5, EE * FF);
  k_gin_mlp<FF, 5><<<cdiv(NN, 64), 256, 0, stream>>>(
      x, FF, agg, c1_eps, c1W1s, c1_b1, c1W2s, c1_b2,
      c1_g, c1_be, c1_m, c1_v, node_emb + 0, DD);

  // 3) GIN layers 2..3 (F_in=64, input = previous slice of node_emb)
  for (int l = 0; l < 2; ++l) {
    const float* in = node_emb + l * HH;
    float* o = node_emb + (l + 1) * HH;
    k_zero<<<cdiv(NN * HH, 256), 256, 0, stream>>>(agg, NN * HH);
    k_edge_scatter<<<cdiv(EE * HH, 256), 256, 0, stream>>>(e_src, e_dst, in, DD, agg, 6, EE * HH);
    k_gin_mlp<HH, 6><<<cdiv(NN, 64), 256, 0, stream>>>(
        in, DD, agg, cs_eps + l,
        csW1s + l * HH * HH, cs_b1 + l * HH,
        csW2s + l * HH * HH, cs_b2 + l * HH,
        cs_g + l * HH, cs_be + l * HH, cs_m + l * HH, cs_v + l * HH, o, DD);
  }

  // 4) mean pool -> classification head (writes out[0:2560] + comp_idx)
  k_zero<<<cdiv(GG * DD, 256), 256, 0, stream>>>(gsum, GG * DD);
  k_zero<<<cdiv(GG, 256), 256, 0, stream>>>(gcnt, GG);
  k_pool_scatter<<<cdiv(NN * DD, 256), 256, 0, stream>>>(node_emb, batch, gsum, gcnt, NN * DD);
  k_pool_final<<<cdiv(GG * DD, 256), 256, 0, stream>>>(gsum, gcnt, gemb);
  k_class_head<<<GG, 128, 0, stream>>>(gemb, nc_W1, nc_b1, nc_W2, nc_b2,
                                       nc_W3, nc_b3, nc_W4, nc_b4, out, cidx);

  // 5) link prediction (writes out[2560:])
  k_comp_contrib<<<NCLS, 128, 0, stream>>>(comp_emb, ep_W1, ep_b1, ccontrib);
  k_link<<<cdiv(CC, 128), 256, 0, stream>>>(cand, batch, cidx, node_emb, ccontrib,
                                            epW1s, epW2s, ep_b2, ep_W3, ep_b3, out);

  (void)in_sizes; (void)n_in; (void)out_size; (void)ws_size;
}
